// TemporalAutoencoder_2491081031720
// MI455X (gfx1250) — matrix operand order repaired and checked by reading, not compiled, not run on hardware
//
#include <hip/hip_runtime.h>
#include <hip/hip_bf16.h>

// Model dims (from reference)
#define BQ    16
#define SQ    256
#define DM    12     // D_MODEL
#define DST   128    // D_STATE
#define HD    12     // HEADDIM
#define NL    4      // N_LAYER
#define VOC   95
#define DI    24     // D_INNER
#define NH    2      // NHEADS
#define DCONV 4
#define CDIM  280    // CONV_DIM
#define DPROJ 306    // D_IN_PROJ
#define EPSF  1e-5f

typedef _Float16 v16h __attribute__((ext_vector_type(16)));
typedef _Float16 v8h  __attribute__((ext_vector_type(8)));
typedef float    v8f  __attribute__((ext_vector_type(8)));

__device__ __forceinline__ v8f wmma_f16(v16h a, v16h b, v8f c) {
  // D = A(16x32 f16) * B(32x16 f16) + C(16x16 f32)
  return __builtin_amdgcn_wmma_f32_16x16x32_f16(false, a, false, b, (short)0, c,
                                                false, false);
}

__device__ __forceinline__ float wred32(float v) {
#pragma unroll
  for (int o = 16; o > 0; o >>= 1) v += __shfl_xor(v, o, 32);
  return v;
}

// ---- A-fragment builders -------------------------------------------------
// ISA 7.12.2 16-bit A layout: lane L holds row M=L%16; half j of the v16h is
// K = (j<8 ? j : j+8) + 8*(L>>4).  We replicate the single useful row into all
// 16 M-rows (rows 1..15 of D are never read, so no masking is needed).

// From a zero-padded 32-half f16 row in LDS: two contiguous ds_load_b128.
__device__ __forceinline__ v16h a_frag_ld16(const _Float16* v) {
  const int hi = (threadIdx.x >> 4) & 1;
  v8h q0 = *(const v8h*)(v + 8 * hi);        // K = 8*hi .. 8*hi+7
  v8h q1 = *(const v8h*)(v + 16 + 8 * hi);   // K = 16+8*hi .. 16+8*hi+7
  v16h a;
#pragma unroll
  for (int j = 0; j < 8; ++j) { a[j] = q0[j]; a[j + 8] = q1[j]; }
  return a;
}

// From 12 f32 register values, scaled by inv (K=12: halves 8..15 are zero).
__device__ __forceinline__ v16h a_frag_k12(const float* h, float inv) {
  const int hi = (threadIdx.x >> 4) & 1;
  v16h a;
#pragma unroll
  for (int j = 0; j < 8; ++j) {
    float lo = h[j];
    float up = (j < 4) ? h[j + 8] : 0.f;
    a[j]     = (_Float16)((hi ? up : lo) * inv);
    a[j + 8] = (_Float16)0.f;
  }
  return a;
}

// From 24 f32 register values, scaled by inv (K=24).
__device__ __forceinline__ v16h a_frag_k24(const float* g, float inv) {
  const int hi = (threadIdx.x >> 4) & 1;
  v16h a;
#pragma unroll
  for (int j = 0; j < 8; ++j) {
    a[j]     = (_Float16)((hi ? g[j + 8] : g[j]) * inv);
    a[j + 8] = (_Float16)((hi ? 0.f : g[j + 8 + 8]) * inv);  // K=j+8 (16..23) or 0
  }
  return a;
}

// B-fragment (32x16 f16): lane L holds N=L%16; half j is K = (j<8 ? j : j+8) + 8*(L>>4).
template <typename F>
__device__ __forceinline__ v16h make_bfrag(F&& f) {
  const int lane = threadIdx.x & 31;
  const int n = lane & 15, hi = lane >> 4;
  v16h r;
#pragma unroll
  for (int j = 0; j < 16; ++j) r[j] = (_Float16)f((j < 8 ? j : j + 8) + 8 * hi, n);
  return r;
}

__device__ __forceinline__ void load12(const float* p, float* h) {
  const float4* q = (const float4*)p;
  float4 a = q[0], b = q[1], c = q[2];
  h[0]=a.x; h[1]=a.y; h[2]=a.z;  h[3]=a.w;
  h[4]=b.x; h[5]=b.y; h[6]=b.z;  h[7]=b.w;
  h[8]=c.x; h[9]=c.y; h[10]=c.z; h[11]=c.w;
}
__device__ __forceinline__ void load24(const float* p, float* h) {
  load12(p, h); load12(p + 12, h + 12);
}

__device__ __forceinline__ float softplusf(float xv) {
  return (xv > 20.f) ? xv : log1pf(__expf(xv));
}
__device__ __forceinline__ float siluf(float xv) {
  return xv / (1.f + __expf(-xv));
}

__global__ __launch_bounds__(512)
void mamba2_autoreg_kernel(const float* __restrict__ x,
                           const float* __restrict__ W_emb,
                           const float* __restrict__ W_in,
                           const float* __restrict__ conv_w,
                           const float* __restrict__ conv_b,
                           const float* __restrict__ dt_bias,
                           const float* __restrict__ A_log,
                           const float* __restrict__ D_param,
                           const float* __restrict__ norm_w,
                           const float* __restrict__ W_out,
                           const float* __restrict__ prenorm_w,
                           const float* __restrict__ normf_w,
                           float* __restrict__ out)
{
  const int b   = blockIdx.x;   // one independent autoregressive chain per WG
  const int tid = threadIdx.x;
  const int wid = tid >> 5;     // 16 wave32 waves
  const int ln  = tid & 31;

  __shared__ __align__(16) _Float16 s_u16[96];   // token vector, f16, zero-padded
  __shared__ __align__(16) float s_h[16];        // residual stream h (12)
  __shared__ __align__(16) float s_zx[320];      // zxbcdt (306 + pad)
  __shared__ __align__(16) float s_xbc[288];     // post-conv silu xBC (280 + pad)
  __shared__ __align__(16) float s_y[32];        // raw y (24)
  __shared__ float s_dt[NH], s_dA[NH];
  __shared__ float s_dtb[NL * NH], s_nA[NL * NH], s_dpar[NL * NH];

  // ---------------- startup ----------------
  for (int i = tid; i < NL * NH; i += blockDim.x) {
    s_dtb[i]  = dt_bias[i];
    s_nA[i]   = -__expf(A_log[i]);
    s_dpar[i] = D_param[i];
  }
  for (int i = tid; i < 96; i += blockDim.x)
    s_u16[i] = (i < VOC) ? (_Float16)x[((size_t)b * SQ + 0) * VOC + i]
                         : (_Float16)0.f;       // u_0 = x[:,0]

  // Resident B-fragments; RMSNorm weight vectors are folded into B at startup.
  v16h bwin0[NL];          // in_proj tile 'wid'    (all 16 waves)   [prenorm_w folded]
  v16h bwin1[NL] = {};     // in_proj tile '16+wid' (waves 0..3)
  v16h bwout[NL] = {};     // out_proj              (wave 0)         [norm_w folded]
  v16h bemb[3]   = {};     // embedding K-chunks    (wave 4)
  v16h blog      = {};     // logits tile 'wid-8'   (waves 8..13)    [normf_w folded]

#pragma unroll
  for (int l = 0; l < NL; ++l) {
    const int colbase = wid * 16;
    bwin0[l] = make_bfrag([&](int k, int n) {
      int col = colbase + n;
      return (k < DM && col < DPROJ)
                 ? prenorm_w[l * DM + k] * W_in[(l * DM + k) * DPROJ + col] : 0.f;
    });
  }
  if (wid < 4) {
#pragma unroll
    for (int l = 0; l < NL; ++l) {
      const int colbase = (16 + wid) * 16;
      bwin1[l] = make_bfrag([&](int k, int n) {
        int col = colbase + n;
        return (k < DM && col < DPROJ)
                   ? prenorm_w[l * DM + k] * W_in[(l * DM + k) * DPROJ + col] : 0.f;
      });
    }
  }
  if (wid == 0) {
#pragma unroll
    for (int l = 0; l < NL; ++l)
      bwout[l] = make_bfrag([&](int k, int n) {
        return (k < DI && n < DM)
                   ? norm_w[l * DI + k] * W_out[(l * DI + k) * DM + n] : 0.f;
      });
  }
  if (wid == 4) {
#pragma unroll
    for (int c = 0; c < 3; ++c)
      bemb[c] = make_bfrag([&](int k, int n) {
        int kg = c * 32 + k;
        return (kg < VOC && n < DM) ? W_emb[kg * DM + n] : 0.f;
      });
  }
  if (wid >= 8 && wid < 14) {
    const int tile = wid - 8;
    blog = make_bfrag([&](int k, int n) {
      int col = tile * 16 + n;
      return (k < DM && col < VOC) ? normf_w[k] * W_emb[col * DM + k] : 0.f;
    });
  }

  // Depthwise-conv taps + weights in registers (channel = tid).
  float cwr[NL][DCONV] = {};
  float cbr[NL] = {};
  float cr[NL][DCONV] = {};
  if (tid < CDIM) {
#pragma unroll
    for (int l = 0; l < NL; ++l) {
#pragma unroll
      for (int j = 0; j < DCONV; ++j) cwr[l][j] = conv_w[(l * CDIM + tid) * DCONV + j];
      cbr[l] = conv_b[l * CDIM + tid];
    }
  }

  // SSM state in VGPRs: wave owns (h = wid>>3, p in {2*(wid&7), +1});
  // lane owns n in [4*ln, 4*ln+4).
  float ss[NL][2][4] = {};
  const bool ssm_wave = ((wid & 7) < 6);
  const int  ssm_h    = wid >> 3;
  const int  ssm_p0   = (wid & 7) * 2;

  __syncthreads();

  // ================= serial autoregressive scan =================
  for (int t = 0; t < SQ; ++t) {
    // ---- embedding: h = u @ W_emb (K chained over 3 WMMAs; A = 2x ds_load_b128)
    if (wid == 4) {
      v8f acc = {};
#pragma unroll
      for (int c = 0; c < 3; ++c) {
        v16h a = a_frag_ld16(s_u16 + 32 * c);
        acc = wmma_f16(a, bemb[c], acc);
      }
      if (ln < DM) s_h[ln] = acc[0];   // C layout: elem0, lanes 0..15 -> row M=0
    }
    __syncthreads();

#pragma unroll
    for (int l = 0; l < NL; ++l) {
      // ---- in_proj with inline RMSNorm: every wave rebuilds A from s_h ----
      {
        float h[12];
        load12(s_h, h);
        float ssq = 0.f;
#pragma unroll
        for (int i = 0; i < 12; ++i) ssq += h[i] * h[i];
        float inv = rsqrtf(ssq * (1.f / DM) + EPSF);
        v16h a = a_frag_k12(h, inv);
        v8f c0 = {};
        c0 = wmma_f16(a, bwin0[l], c0);
        if (ln < 16) s_zx[wid * 16 + ln] = c0[0];
        if (wid < 4) {
          v8f c1 = {};
          c1 = wmma_f16(a, bwin1[l], c1);
          if (ln < 16) s_zx[(16 + wid) * 16 + ln] = c1[0];
        }
      }
      __syncthreads();

      // ---- depthwise conv (shift regs) + silu; dt -> softplus, dA ----
      if (tid < CDIM) {
        float cur = s_zx[DI + tid];
        cr[l][0] = cr[l][1]; cr[l][1] = cr[l][2]; cr[l][2] = cr[l][3]; cr[l][3] = cur;
        float acc = cbr[l];
#pragma unroll
        for (int j = 0; j < DCONV; ++j) acc += cr[l][j] * cwr[l][j];
        s_xbc[tid] = siluf(acc);
      } else if (tid < CDIM + NH) {
        int hh = tid - CDIM;
        float draw = s_zx[DI + CDIM + hh];               // cols 304,305
        float dtv  = softplusf(draw + s_dtb[l * NH + hh]);
        s_dt[hh] = dtv;
        s_dA[hh] = __expf(s_nA[l * NH + hh] * dtv);
      }
      __syncthreads();

      // ---- SSM: ss = ss*dA + dt*B*x ; y = ss . C + D*x (state in VGPRs) ----
      if (ssm_wave) {
        const float dtv = s_dt[ssm_h];
        const float dA  = s_dA[ssm_h];
        const float4 Bm = *(const float4*)&s_xbc[DI + 4 * ln];        // B[n0..n0+3]
        const float4 Cm = *(const float4*)&s_xbc[DI + DST + 4 * ln];  // C[n0..n0+3]
        const float k0 = dtv * Bm.x, k1 = dtv * Bm.y, k2 = dtv * Bm.z, k3 = dtv * Bm.w;
#pragma unroll
        for (int dp = 0; dp < 2; ++dp) {
          const int p = ssm_p0 + dp;
          const float xv = s_xbc[ssm_h * HD + p];
          float s0 = ss[l][dp][0] * dA + k0 * xv;
          float s1 = ss[l][dp][1] * dA + k1 * xv;
          float s2 = ss[l][dp][2] * dA + k2 * xv;
          float s3 = ss[l][dp][3] * dA + k3 * xv;
          ss[l][dp][0] = s0; ss[l][dp][1] = s1; ss[l][dp][2] = s2; ss[l][dp][3] = s3;
          float part = s0 * Cm.x + s1 * Cm.y + s2 * Cm.z + s3 * Cm.w;
          part = wred32(part);  // reduce over D_STATE (32 lanes x 4 n each)
          if (ln == 0) s_y[ssm_h * HD + p] = part + s_dpar[l * NH + ssm_h] * xv;
        }
      }
      __syncthreads();

      // ---- out_proj with inline gate+RMSNorm: h += rmsnorm(y*silu(z)) @ W_out'
      if (wid == 0) {
        float y[24], z[24], g[24];
        load24(s_y, y);
        load24(s_zx, z);                  // z = zxbcdt[:, :24]
        float ssq = 0.f;
#pragma unroll
        for (int i = 0; i < 24; ++i) { g[i] = y[i] * siluf(z[i]); ssq += g[i] * g[i]; }
        float inv = rsqrtf(ssq * (1.f / DI) + EPSF);
        v16h a = a_frag_k24(g, inv);
        v8f c = {};
        c = wmma_f16(a, bwout[l], c);
        if (ln < DM) s_h[ln] += c[0];
      }
      __syncthreads();
    }  // layers

    // ---- logits with inline final RMSNorm; emit + feed back as next token ----
    if (wid >= 8 && wid < 14) {
      float h[12];
      load12(s_h, h);
      float ssq = 0.f;
#pragma unroll
      for (int i = 0; i < 12; ++i) ssq += h[i] * h[i];
      float inv = rsqrtf(ssq * (1.f / DM) + EPSF);
      v16h a = a_frag_k12(h, inv);
      v8f c = {};
      c = wmma_f16(a, blog, c);
      if (ln < 16) {
        int col = (wid - 8) * 16 + ln;
        if (col < VOC) {
          float val = c[0];
          out[((size_t)b * SQ + t) * VOC + col] = val;
          s_u16[col] = (_Float16)val;     // autoregressive feedback
        }
      }
    }
    __syncthreads();
  }
}

extern "C" void kernel_launch(void* const* d_in, const int* in_sizes, int n_in,
                              void* d_out, int out_size, void* d_ws, size_t ws_size,
                              hipStream_t stream) {
  (void)in_sizes; (void)n_in; (void)out_size; (void)d_ws; (void)ws_size;
  const float* x    = (const float*)d_in[0];
  const float* Wemb = (const float*)d_in[1];
  const float* Win  = (const float*)d_in[2];
  const float* cw   = (const float*)d_in[3];
  const float* cb   = (const float*)d_in[4];
  const float* dtb  = (const float*)d_in[5];
  const float* alog = (const float*)d_in[6];
  const float* dpar = (const float*)d_in[7];
  const float* nw   = (const float*)d_in[8];
  const float* Wout = (const float*)d_in[9];
  const float* prew = (const float*)d_in[10];
  const float* fw   = (const float*)d_in[11];
  mamba2_autoreg_kernel<<<dim3(BQ), dim3(512), 0, stream>>>(
      x, Wemb, Win, cw, cb, dtb, alog, dpar, nw, Wout, prew, fw, (float*)d_out);
}
